// acModel_32787780338139
// MI455X (gfx1250) — compile-verified, matrix operand order checked by reading, not run on hardware
//
#include <hip/hip_runtime.h>
#include <hip/hip_bf16.h>

// ---------------------------------------------------------------------------
// 3-layer LSTM rollout (H=1024, B=256, S=128, D=63) for gfx1250 (MI455X).
// GEMMs use v_wmma_f32_16x16x32_bf16 (bf16 A/B, fp32 accumulate).
// Wave tile 32x64 = 2x4 WMMA tiles -> 8 WMMAs per 6 fragment loads.
// __launch_bounds__(256,1): grid is launch-limited (512 waves/GEMM), so give
// each wave the full VGPR budget to avoid accumulator spills in the K-loop.
// Weights pre-converted to bf16 once per call; all weights stay L2-resident.
// ---------------------------------------------------------------------------

#define HSZ 1024
#define BSZ 256
#define SEQ 128
#define DSZ 63
#define DPAD 64
#define G4H 4096

typedef __attribute__((ext_vector_type(16))) __bf16 v16bf;
typedef __attribute__((ext_vector_type(8)))  float  v8f;

union Frag { v16bf v; uint4 q[2]; };

__device__ __forceinline__ unsigned short f2bf(float f) {
    union { float f; unsigned u; } x; x.f = f;
    unsigned r = x.u + 0x7FFFu + ((x.u >> 16) & 1u);   // round-to-nearest-even
    return (unsigned short)(r >> 16);
}

// Load a 16x32 bf16 A/B fragment for lane (lane16, hi) from a row-major
// [rows][ld] bf16 matrix. Row chunk of K=32 bf16 = 4 uint4; the gfx1250
// 16-bit WMMA lane layout maps to uint4 indices {hi, 2+hi} of that chunk.
__device__ __forceinline__ v16bf load_frag(const uint4* rowchunk, int hi) {
    Frag f;
    f.q[0] = rowchunk[hi];
    f.q[1] = rowchunk[2 + hi];
    return f.v;
}

// Accumulate a 32x64 output tile over K: acc[2][4] += A[32 rows] @ W[64 rows]^T
__device__ __forceinline__ void mma_accum(const unsigned short* __restrict__ A, int lda,
                                          const unsigned short* __restrict__ W, int ldw,
                                          int K, int m0, int n0, int lane16, int hi,
                                          v8f acc[2][4], bool do_prefetch)
{
    const uint4* ar0 = (const uint4*)(A + (size_t)(m0 + lane16)      * lda);
    const uint4* ar1 = (const uint4*)(A + (size_t)(m0 + 16 + lane16) * lda);
    const uint4* wr0 = (const uint4*)(W + (size_t)(n0 + lane16)      * ldw);
    const uint4* wr1 = (const uint4*)(W + (size_t)(n0 + 16 + lane16) * ldw);
    const uint4* wr2 = (const uint4*)(W + (size_t)(n0 + 32 + lane16) * ldw);
    const uint4* wr3 = (const uint4*)(W + (size_t)(n0 + 48 + lane16) * ldw);

    for (int k = 0; k < K; k += 32) {
        const int b = k >> 3;                          // uint4 index of K-chunk
        if (do_prefetch && (k + 32 < K)) {             // emits global_prefetch_b8
            __builtin_prefetch(wr0 + b + 4, 0, 1);
            __builtin_prefetch(wr2 + b + 4, 0, 1);
        }
        v16bf a0 = load_frag(ar0 + b, hi);
        v16bf a1 = load_frag(ar1 + b, hi);
        v16bf b0 = load_frag(wr0 + b, hi);
        v16bf b1 = load_frag(wr1 + b, hi);
        v16bf b2 = load_frag(wr2 + b, hi);
        v16bf b3 = load_frag(wr3 + b, hi);
        acc[0][0] = __builtin_amdgcn_wmma_f32_16x16x32_bf16(false, a0, false, b0, (short)0, acc[0][0], false, false);
        acc[0][1] = __builtin_amdgcn_wmma_f32_16x16x32_bf16(false, a0, false, b1, (short)0, acc[0][1], false, false);
        acc[0][2] = __builtin_amdgcn_wmma_f32_16x16x32_bf16(false, a0, false, b2, (short)0, acc[0][2], false, false);
        acc[0][3] = __builtin_amdgcn_wmma_f32_16x16x32_bf16(false, a0, false, b3, (short)0, acc[0][3], false, false);
        acc[1][0] = __builtin_amdgcn_wmma_f32_16x16x32_bf16(false, a1, false, b0, (short)0, acc[1][0], false, false);
        acc[1][1] = __builtin_amdgcn_wmma_f32_16x16x32_bf16(false, a1, false, b1, (short)0, acc[1][1], false, false);
        acc[1][2] = __builtin_amdgcn_wmma_f32_16x16x32_bf16(false, a1, false, b2, (short)0, acc[1][2], false, false);
        acc[1][3] = __builtin_amdgcn_wmma_f32_16x16x32_bf16(false, a1, false, b3, (short)0, acc[1][3], false, false);
    }
}

// ---------------------------------------------------------------------------
// gates[M=256][N=4096] = X[M][Kx] @ Wx[N][Kx]^T + H[M][1024] @ Wh[N][1024]^T + bias
// Block: 256 threads = 8 waves, arranged 2(M) x 4(N); wave tile 32x64.
// Block tile 64x256.  Grid: (4096/256, 256/64) = (16, 4).
// ---------------------------------------------------------------------------
__global__ void __launch_bounds__(256, 1)
gemm_gates_kernel(const unsigned short* __restrict__ X, int ldx, int Kx,
                  const unsigned short* __restrict__ Wx,
                  const unsigned short* __restrict__ Hm,
                  const unsigned short* __restrict__ Wh,
                  const float* __restrict__ bias,
                  float* __restrict__ gates)
{
    const int tid    = threadIdx.x;
    const int lane   = tid & 31;
    const int w      = tid >> 5;
    const int lane16 = lane & 15;
    const int hi     = lane >> 4;
    const int wm     = w >> 2;          // 0..1
    const int wn     = w & 3;           // 0..3
    const int m0     = blockIdx.y * 64 + wm * 32;
    const int n0     = blockIdx.x * 256 + wn * 64;

    v8f acc[2][4] = {};

    mma_accum(X,  ldx, Wx, Kx,  Kx,  m0, n0, lane16, hi, acc, false);  // X @ Wx^T
    mma_accum(Hm, HSZ, Wh, HSZ, HSZ, m0, n0, lane16, hi, acc, true);   // H @ Wh^T

    // ---- epilogue: D layout -> gates (+bias) ----
    #pragma unroll
    for (int i = 0; i < 2; ++i) {
        #pragma unroll
        for (int j = 0; j < 4; ++j) {
            const int col = n0 + j * 16 + lane16;
            const float bval = bias[col];
            #pragma unroll
            for (int r = 0; r < 8; ++r) {
                const int m = m0 + i * 16 + hi * 8 + r;
                gates[(size_t)m * G4H + col] = acc[i][j][r] + bval;
            }
        }
    }
}

// ---------------------------------------------------------------------------
// Pointwise LSTM cell: c_new = sig(f)*c + sig(i)*tanh(g); h = sig(o)*tanh(c_new)
// ---------------------------------------------------------------------------
__global__ void lstm_cell_kernel(const float* __restrict__ gates,
                                 float* __restrict__ c,
                                 unsigned short* __restrict__ h)
{
    const int i = blockIdx.x * blockDim.x + threadIdx.x;   // 0 .. B*H-1
    if (i >= BSZ * HSZ) return;
    const int b = i >> 10, u = i & 1023;
    const float* g = gates + (size_t)b * G4H;
    const float ig = g[u], fg = g[u + 1024], gg = g[u + 2048], og = g[u + 3072];
    const float si = 1.f / (1.f + __expf(-ig));
    const float sf = 1.f / (1.f + __expf(-fg));
    const float so = 1.f / (1.f + __expf(-og));
    const float cn = sf * c[i] + si * tanhf(gg);
    c[i] = cn;
    h[i] = f2bf(so * tanhf(cn));
}

// ---------------------------------------------------------------------------
// Projection: out[:, t, :] = h3[256][1024] @ Wp[64][1024]^T + b_proj  (N eff 63)
// Block: 8 waves arranged 4(M) x 2(N); block tile 64x64. Grid: 4 blocks.
// ---------------------------------------------------------------------------
__global__ void __launch_bounds__(256, 1)
proj_kernel(const unsigned short* __restrict__ H3,
            const unsigned short* __restrict__ Wp,
            const float* __restrict__ bproj,
            float* __restrict__ out, int t)
{
    const int tid    = threadIdx.x;
    const int lane   = tid & 31;
    const int w      = tid >> 5;
    const int lane16 = lane & 15;
    const int hi     = lane >> 4;
    const int wm     = w >> 1;          // 0..3
    const int wn     = w & 1;           // 0..1
    const int m0     = blockIdx.x * 64 + wm * 16;
    const int n0     = wn * 32;

    v8f acc0 = {}; v8f acc1 = {};
    const uint4* hr  = (const uint4*)(H3 + (size_t)(m0 + lane16)      * HSZ);
    const uint4* w0r = (const uint4*)(Wp + (size_t)(n0 + lane16)      * HSZ);
    const uint4* w1r = (const uint4*)(Wp + (size_t)(n0 + 16 + lane16) * HSZ);
    for (int k = 0; k < HSZ; k += 32) {
        const int b = k >> 3;
        v16bf a  = load_frag(hr  + b, hi);
        v16bf b0 = load_frag(w0r + b, hi);
        v16bf b1 = load_frag(w1r + b, hi);
        acc0 = __builtin_amdgcn_wmma_f32_16x16x32_bf16(false, a, false, b0,
                                                       (short)0, acc0, false, false);
        acc1 = __builtin_amdgcn_wmma_f32_16x16x32_bf16(false, a, false, b1,
                                                       (short)0, acc1, false, false);
    }
    const int c0 = n0 + lane16, c1 = n0 + 16 + lane16;
    #pragma unroll
    for (int r = 0; r < 8; ++r) {
        const int m = m0 + hi * 8 + r;
        float* orow = out + ((size_t)m * SEQ + t) * DSZ;
        if (c0 < DSZ) orow[c0] = acc0[r] + bproj[c0];
        if (c1 < DSZ) orow[c1] = acc1[r] + bproj[c1];
    }
}

// ---------------------------------------------------------------------------
// Input select: X[b][0..63] = bf16( use_real ? seq[b,t,:] : out[b,t-1,:] ), pad d=63
// ---------------------------------------------------------------------------
__global__ void select_input_kernel(const float* __restrict__ seq,
                                    const float* __restrict__ outbuf,
                                    unsigned short* __restrict__ X,
                                    const int* __restrict__ cond_len,
                                    const int* __restrict__ gt_len, int t)
{
    const int i = blockIdx.x * blockDim.x + threadIdx.x;   // 0 .. 256*64-1
    if (i >= BSZ * DPAD) return;
    const int b = i >> 6, d = i & 63;
    const int gt = *gt_len, cl = *cond_len;
    int period = gt + cl; if (period <= 0) period = 1;
    const bool use_real = (t % period) < gt;
    float v = 0.f;
    if (d < DSZ) {
        if (use_real)      v = seq[((size_t)b * SEQ + t) * DSZ + d];
        else if (t > 0)    v = outbuf[((size_t)b * SEQ + (t - 1)) * DSZ + d];
        // t==0 non-real: reference prev_out init is zeros -> v stays 0
    }
    X[(size_t)b * DPAD + d] = f2bf(v);
}

// ---------------------------------------------------------------------------
// Weight fp32 -> bf16 with N/K zero-padding. dst is [Npad][Kpad].
// ---------------------------------------------------------------------------
__global__ void convert_pad_kernel(const float* __restrict__ src,
                                   unsigned short* __restrict__ dst,
                                   int Nin, int Kin, int Npad, int Kpad)
{
    const int i = blockIdx.x * blockDim.x + threadIdx.x;
    if (i >= Npad * Kpad) return;
    const int n = i / Kpad, k = i - n * Kpad;
    const float v = (n < Nin && k < Kin) ? src[(size_t)n * Kin + k] : 0.f;
    dst[i] = f2bf(v);
}

__global__ void bias_combine_kernel(const float* __restrict__ bih,
                                    const float* __restrict__ bhh,
                                    float* __restrict__ dst)
{
    const int i = blockIdx.x * blockDim.x + threadIdx.x;
    if (i < G4H) dst[i] = bih[i] + bhh[i];
}

__global__ void zero_state_kernel(unsigned short* h1, unsigned short* h2, unsigned short* h3,
                                  float* c1, float* c2, float* c3)
{
    const int i = blockIdx.x * blockDim.x + threadIdx.x;
    if (i >= BSZ * HSZ) return;
    h1[i] = 0; h2[i] = 0; h3[i] = 0;
    c1[i] = 0.f; c2[i] = 0.f; c3[i] = 0.f;
}

// ---------------------------------------------------------------------------
extern "C" void kernel_launch(void* const* d_in, const int* in_sizes, int n_in,
                              void* d_out, int out_size, void* d_ws, size_t ws_size,
                              hipStream_t stream)
{
    const float* real_seq = (const float*)d_in[0];
    const float* W_ih1 = (const float*)d_in[1];  const float* W_hh1 = (const float*)d_in[2];
    const float* b_ih1 = (const float*)d_in[3];  const float* b_hh1 = (const float*)d_in[4];
    const float* W_ih2 = (const float*)d_in[5];  const float* W_hh2 = (const float*)d_in[6];
    const float* b_ih2 = (const float*)d_in[7];  const float* b_hh2 = (const float*)d_in[8];
    const float* W_ih3 = (const float*)d_in[9];  const float* W_hh3 = (const float*)d_in[10];
    const float* b_ih3 = (const float*)d_in[11]; const float* b_hh3 = (const float*)d_in[12];
    const float* W_proj = (const float*)d_in[13]; const float* b_proj = (const float*)d_in[14];
    const int* cond_len = (const int*)d_in[15];
    const int* gt_len   = (const int*)d_in[16];
    float* out = (float*)d_out;

    // ---- carve workspace (bf16 buffers as ushort) ----
    char* base = (char*)d_ws;
    size_t cur = 0;
    auto alloc = [&](size_t bytes) -> char* {
        char* p = base + cur;
        cur = (cur + bytes + 255) & ~(size_t)255;
        return p;
    };
    unsigned short* Xbf  = (unsigned short*)alloc((size_t)BSZ * DPAD * 2);
    unsigned short* Wi1  = (unsigned short*)alloc((size_t)G4H * DPAD * 2);
    unsigned short* Wh1  = (unsigned short*)alloc((size_t)G4H * HSZ * 2);
    unsigned short* Wi2  = (unsigned short*)alloc((size_t)G4H * HSZ * 2);
    unsigned short* Wh2  = (unsigned short*)alloc((size_t)G4H * HSZ * 2);
    unsigned short* Wi3  = (unsigned short*)alloc((size_t)G4H * HSZ * 2);
    unsigned short* Wh3  = (unsigned short*)alloc((size_t)G4H * HSZ * 2);
    unsigned short* Wp   = (unsigned short*)alloc((size_t)DPAD * HSZ * 2);
    float* bias1 = (float*)alloc((size_t)G4H * 4);
    float* bias2 = (float*)alloc((size_t)G4H * 4);
    float* bias3 = (float*)alloc((size_t)G4H * 4);
    float* gates = (float*)alloc((size_t)BSZ * G4H * 4);
    unsigned short* h1 = (unsigned short*)alloc((size_t)BSZ * HSZ * 2);
    unsigned short* h2 = (unsigned short*)alloc((size_t)BSZ * HSZ * 2);
    unsigned short* h3 = (unsigned short*)alloc((size_t)BSZ * HSZ * 2);
    float* c1 = (float*)alloc((size_t)BSZ * HSZ * 4);
    float* c2 = (float*)alloc((size_t)BSZ * HSZ * 4);
    float* c3 = (float*)alloc((size_t)BSZ * HSZ * 4);
    (void)ws_size; (void)in_sizes; (void)n_in; (void)out_size;

    // ---- one-time per call: convert weights, combine biases, zero state ----
    const int T = 256;
    {
        int n;
        n = G4H * DPAD;
        convert_pad_kernel<<<(n + T - 1) / T, T, 0, stream>>>(W_ih1, Wi1, G4H, DSZ, G4H, DPAD);
        n = G4H * HSZ;
        convert_pad_kernel<<<(n + T - 1) / T, T, 0, stream>>>(W_hh1, Wh1, G4H, HSZ, G4H, HSZ);
        convert_pad_kernel<<<(n + T - 1) / T, T, 0, stream>>>(W_ih2, Wi2, G4H, HSZ, G4H, HSZ);
        convert_pad_kernel<<<(n + T - 1) / T, T, 0, stream>>>(W_hh2, Wh2, G4H, HSZ, G4H, HSZ);
        convert_pad_kernel<<<(n + T - 1) / T, T, 0, stream>>>(W_ih3, Wi3, G4H, HSZ, G4H, HSZ);
        convert_pad_kernel<<<(n + T - 1) / T, T, 0, stream>>>(W_hh3, Wh3, G4H, HSZ, G4H, HSZ);
        n = DPAD * HSZ;
        convert_pad_kernel<<<(n + T - 1) / T, T, 0, stream>>>(W_proj, Wp, DSZ, HSZ, DPAD, HSZ);
        bias_combine_kernel<<<(G4H + T - 1) / T, T, 0, stream>>>(b_ih1, b_hh1, bias1);
        bias_combine_kernel<<<(G4H + T - 1) / T, T, 0, stream>>>(b_ih2, b_hh2, bias2);
        bias_combine_kernel<<<(G4H + T - 1) / T, T, 0, stream>>>(b_ih3, b_hh3, bias3);
        n = BSZ * HSZ;
        zero_state_kernel<<<(n + T - 1) / T, T, 0, stream>>>(h1, h2, h3, c1, c2, c3);
    }

    const dim3 gGrid(G4H / 256, BSZ / 64);   // (16, 4) blocks, 8 waves each
    const int  cellBlocks = (BSZ * HSZ + T - 1) / T;

    for (int t = 0; t < SEQ; ++t) {
        select_input_kernel<<<(BSZ * DPAD + T - 1) / T, T, 0, stream>>>(
            real_seq, out, Xbf, cond_len, gt_len, t);

        // layer 1
        gemm_gates_kernel<<<gGrid, 256, 0, stream>>>(Xbf, DPAD, DPAD, Wi1, h1, Wh1, bias1, gates);
        lstm_cell_kernel<<<cellBlocks, T, 0, stream>>>(gates, c1, h1);
        // layer 2
        gemm_gates_kernel<<<gGrid, 256, 0, stream>>>(h1, HSZ, HSZ, Wi2, h2, Wh2, bias2, gates);
        lstm_cell_kernel<<<cellBlocks, T, 0, stream>>>(gates, c2, h2);
        // layer 3
        gemm_gates_kernel<<<gGrid, 256, 0, stream>>>(h2, HSZ, HSZ, Wi3, h3, Wh3, bias3, gates);
        lstm_cell_kernel<<<cellBlocks, T, 0, stream>>>(gates, c3, h3);
        // projection -> out[:, t, :]
        proj_kernel<<<4, 256, 0, stream>>>(h3, Wp, b_proj, out, t);
    }
}
